// MetaLinear2_20220706030481
// MI455X (gfx1250) — matrix operand order repaired and checked by reading, not compiled, not run on hardware
//
#include <hip/hip_runtime.h>
#include <hip/hip_bf16.h>

// MetaLinear2 on MI455X (gfx1250, wave32, WMMA).
//
// Shapes: B=4,S=512 -> T=2048 tokens; IN=256, OUT=256, HID=32. EPS=1e-5.
//
// Fused design: the generated weight matrix w[bs,o,i] (512 MB if materialized)
// is produced 16x16 tile at a time inside the v_wmma_f32_16x16x32_f16
// accumulator (M=16 tokens, N=16 i's, K=HID=32 -> exactly one WMMA per tile,
// bias b2 injected via the f32 C operand). LayerNorm stats and the x-contraction
// are reduced straight out of the accumulator registers. HBM traffic ~13 MB vs
// >1 GB naive; total matrix work 8.6 GFLOP of f16 WMMA w/ f32 accumulate.

typedef __attribute__((ext_vector_type(16))) _Float16 v16h;
typedef __attribute__((ext_vector_type(8)))  float    v8f;

#define T_TOK   2048
#define IN_DIM  256
#define OUT_DIM 256
#define HID     32
#define LN_EPS  1e-5f

// ---------------------------------------------------------------------------
// Kernel 1: repack W2 (f32 [OUT*IN, HID]) into f16 WMMA B-fragments.
// Fragment (o, itile): B[k=h, n=i_local] = w2[(o*256 + itile*16 + n)*32 + k].
// B order used here (16-bit 32x16, same as A): lane L -> n = L&15, element j -> k = 16*(L>>4)+j.
// Stored so each lane's 16 halves are contiguous (32 B -> 2x b128 loads).
// ---------------------------------------------------------------------------
__global__ void pack_w2_kernel(const float* __restrict__ w2,
                               _Float16* __restrict__ W2pack) {
    int t    = blockIdx.x * 256 + threadIdx.x;     // 256*16*32 = 131072 frags*lanes
    int lane = t & 31;
    int it   = (t >> 5) & 15;
    int o    = t >> 9;
    int row   = o * IN_DIM + it * 16 + (lane & 15);
    int hbase = 16 * (lane >> 4);
    const float* src = w2 + (size_t)row * HID + hbase;   // 16 consecutive f32
    _Float16*    dst = W2pack + (size_t)t * 16;
#pragma unroll
    for (int j = 0; j < 16; ++j) dst[j] = (_Float16)src[j];
}

// ---------------------------------------------------------------------------
// Kernel 2: per-token setup. One wave (32 lanes) per token.
//  hw = x@W1^T + b1  -> stored f16 directly in A-fragment lane order
//  hb = x@U1^T + c1 ; b_pre = hb@U2^T + c2 ; b_ln = LN(b_pre) (f32)
//  xs = sum_i x[i]
// A layout (16-bit 16x32): lane L -> m = L&15, element j -> k = 16*(L>>4)+j.
// ---------------------------------------------------------------------------
__global__ void token_setup_kernel(const float* __restrict__ x,
                                   const float* __restrict__ w1,
                                   const float* __restrict__ b1,
                                   const float* __restrict__ u1,
                                   const float* __restrict__ c1,
                                   const float* __restrict__ u2,
                                   const float* __restrict__ c2,
                                   _Float16* __restrict__ Apack,
                                   float* __restrict__ bln,
                                   float* __restrict__ xs) {
    __shared__ float xl[8][IN_DIM];
    __shared__ float hbl[8][HID];
    int w  = threadIdx.x >> 5;
    int l  = threadIdx.x & 31;
    int tk = blockIdx.x * 8 + w;

    const float* xr = x + (size_t)tk * IN_DIM;
    float part = 0.f;
#pragma unroll
    for (int k = 0; k < 8; ++k) {
        float v = xr[l + 32 * k];
        xl[w][l + 32 * k] = v;
        part += v;
    }
#pragma unroll
    for (int m = 16; m >= 1; m >>= 1) part += __shfl_xor(part, m, 32);
    if (l == 0) xs[tk] = part;
    __syncthreads();

    // lane l computes hidden channel h = l for both paths
    float hw = b1[l];
    float hb = c1[l];
    for (int i = 0; i < IN_DIM; ++i) {
        float xi = xl[w][i];
        hw = fmaf(w1[l * IN_DIM + i], xi, hw);
        hb = fmaf(u1[l * IN_DIM + i], xi, hb);
    }
    hbl[w][l] = hb;

    // scatter hw[k=l] into its A-fragment slot
    {
        int ttile = tk >> 4, mm = tk & 15, k = l;
        int lp = mm + 16 * (k >> 4);
        int j  = k & 15;
        Apack[(size_t)ttile * 512 + lp * 16 + j] = (_Float16)hw;
    }
    __syncthreads();

    // bias path: b_pre[o] for o = l + 32q, then LayerNorm over 256
    float bp[8], s = 0.f, s2 = 0.f;
#pragma unroll
    for (int q = 0; q < 8; ++q) {
        int o = l + 32 * q;
        float acc = c2[o];
#pragma unroll
        for (int h = 0; h < HID; ++h) acc = fmaf(u2[o * HID + h], hbl[w][h], acc);
        bp[q] = acc;
        s += acc;
        s2 = fmaf(acc, acc, s2);
    }
#pragma unroll
    for (int m = 16; m >= 1; m >>= 1) {
        s  += __shfl_xor(s,  m, 32);
        s2 += __shfl_xor(s2, m, 32);
    }
    float mu  = s * (1.f / 256.f);
    float var = s2 * (1.f / 256.f) - mu * mu;
    float rs  = rsqrtf(fmaxf(var, 0.f) + LN_EPS);
#pragma unroll
    for (int q = 0; q < 8; ++q)
        bln[(size_t)tk * OUT_DIM + l + 32 * q] = (bp[q] - mu) * rs;
}

// ---------------------------------------------------------------------------
// Kernel 3: main fused generator-GEMM + LayerNorm + matvec.
// Grid: 128 WGs (one 16-token tile each), 256 threads = 8 wave32 waves.
// Wave w owns o = 32w .. 32w+31. Per (o, itile): ONE v_wmma_f32_16x16x32_f16
// produces the 16x16 w-tile (with b2 pre-loaded in C); epilogue accumulates
// sum(w), sum(w^2), sum(w*x) per row from the accumulator, then a 16-lane
// xor-butterfly finishes the i-reduction and lane 0 of each half emits y.
// ---------------------------------------------------------------------------
__global__ void meta_mm_kernel(const float* __restrict__ x,
                               const float* __restrict__ b2,
                               const _Float16* __restrict__ W2pack,
                               const _Float16* __restrict__ Apack,
                               const float* __restrict__ bln,
                               const float* __restrict__ xs,
                               float* __restrict__ y) {
    // x transposed with row stride 20 floats: keeps 4-float groups 16B aligned
    // for ds_load_b128 and spreads the 64 LDS banks conflict-free.
    __shared__ alignas(16) float xT[IN_DIM][20];
    __shared__ float ytile[16][OUT_DIM];
    __shared__ float xsl[16];

    int tid   = threadIdx.x;
    int ttile = blockIdx.x;
    int t0    = ttile * 16;

    for (int idx = tid; idx < 16 * IN_DIM; idx += 256) {
        int m = idx >> 8, i = idx & 255;
        xT[i][m] = x[(size_t)(t0 + m) * IN_DIM + i];
    }
    if (tid < 16) xsl[tid] = xs[t0 + tid];
    __syncthreads();

    int w  = tid >> 5, l = tid & 31;
    int li = l & 15,  lh = l >> 4;

    // A fragment (hw tile for these 16 tokens): same for every o this wave does
    v16h af = *(const v16h*)(Apack + (size_t)ttile * 512 + l * 16);

    for (int ok = 0; ok < 32; ++ok) {
        int o = w * 32 + ok;
        float s1[8], s2[8], dot[8];
#pragma unroll
        for (int r = 0; r < 8; ++r) { s1[r] = 0.f; s2[r] = 0.f; dot[r] = 0.f; }

        const _Float16* bbase = W2pack + (size_t)o * 16 * 512 + l * 16;
        const float*    b2row = b2 + (size_t)o * IN_DIM;

#pragma unroll 2
        for (int it = 0; it < 16; ++it) {
            v16h  bf  = *(const v16h*)(bbase + (size_t)it * 512);
            float b2v = b2row[it * 16 + li];           // C[m][n] = b2[o*256+i], all m
            v8f c;
#pragma unroll
            for (int r = 0; r < 8; ++r) c[r] = b2v;
            c = __builtin_amdgcn_wmma_f32_16x16x32_f16(
                    false, af, false, bf, (short)0, c, false, false);

            // x[m][i] for this lane's column i = it*16+li, rows m = lh*8 + r
            const float* xcol = &xT[it * 16 + li][lh * 8];
            float4 xa = *(const float4*)(xcol);
            float4 xb = *(const float4*)(xcol + 4);
            float  xv[8] = {xa.x, xa.y, xa.z, xa.w, xb.x, xb.y, xb.z, xb.w};
#pragma unroll
            for (int r = 0; r < 8; ++r) {
                float wv = c[r];
                s1[r]  += wv;
                s2[r]   = fmaf(wv, wv, s2[r]);
                dot[r]  = fmaf(wv, xv[r], dot[r]);
            }
        }

        // finish i-reduction across the 16 lanes of each half-wave
#pragma unroll
        for (int msk = 1; msk <= 8; msk <<= 1) {
#pragma unroll
            for (int r = 0; r < 8; ++r) {
                s1[r]  += __shfl_xor(s1[r],  msk, 32);
                s2[r]  += __shfl_xor(s2[r],  msk, 32);
                dot[r] += __shfl_xor(dot[r], msk, 32);
            }
        }
        if (li == 0) {
#pragma unroll
            for (int r = 0; r < 8; ++r) {
                int   mrow = lh * 8 + r;
                float mu   = s1[r] * (1.f / 256.f);
                float var  = s2[r] * (1.f / 256.f) - mu * mu;
                float rs   = rsqrtf(fmaxf(var, 0.f) + LN_EPS);
                ytile[mrow][o] = (dot[r] - mu * xsl[mrow]) * rs;
            }
        }
    }
    __syncthreads();

    for (int idx = tid; idx < 16 * OUT_DIM; idx += 256) {
        int m = idx >> 8, o = idx & 255;
        size_t gi = (size_t)(t0 + m) * OUT_DIM + o;
        y[gi] = ytile[m][o] + bln[gi];
    }
}

// ---------------------------------------------------------------------------
extern "C" void kernel_launch(void* const* d_in, const int* in_sizes, int n_in,
                              void* d_out, int out_size, void* d_ws, size_t ws_size,
                              hipStream_t stream) {
    const float* x  = (const float*)d_in[0];
    const float* w1 = (const float*)d_in[1];
    const float* b1 = (const float*)d_in[2];
    const float* w2 = (const float*)d_in[3];
    const float* b2 = (const float*)d_in[4];
    const float* u1 = (const float*)d_in[5];
    const float* c1 = (const float*)d_in[6];
    const float* u2 = (const float*)d_in[7];
    const float* c2 = (const float*)d_in[8];
    float* y = (float*)d_out;

    char* ws = (char*)d_ws;
    // workspace layout (bytes):
    //   W2pack f16 : 256*16*32*16*2 = 4,194,304
    //   Apack  f16 : 128*512*2      =   131,072
    //   bln    f32 : 2048*256*4     = 2,097,152
    //   xs     f32 : 2048*4         =     8,192   (total ~6.43 MB)
    _Float16* W2pack = (_Float16*)ws;
    _Float16* Apack  = (_Float16*)(ws + 4194304);
    float*    bln    = (float*)(ws + 4194304 + 131072);
    float*    xsbuf  = (float*)(ws + 4194304 + 131072 + 2097152);

    pack_w2_kernel<<<512, 256, 0, stream>>>(w2, W2pack);
    token_setup_kernel<<<256, 256, 0, stream>>>(x, w1, b1, u1, c1, u2, c2,
                                                Apack, bln, xsbuf);
    meta_mm_kernel<<<128, 256, 0, stream>>>(x, b2, W2pack, Apack, bln, xsbuf, y);
}
